// DeforAttn_4724464025951
// MI455X (gfx1250) — matrix operand order repaired and checked by reading, not compiled, not run on hardware
//
#include <hip/hip_runtime.h>
#include <stdint.h>

// ---------- types for WMMA ----------
typedef __attribute__((ext_vector_type(16))) __bf16 v16bf;
typedef __attribute__((ext_vector_type(8)))  float  v8f;

#define C_DIM   256
#define HEADS   8
#define POINTS  8
#define LEVELS  2
#define HGRID   128
#define WGRID   128
#define NQ      (HGRID * WGRID)      // 16384
#define DH      (C_DIM / HEADS)      // 32
#define KDIM    256                  // reduction dim for every GEMM here
#define LDSPITCH 264                 // 256 + 8 pad (528B rows: 16B aligned, bank-skewed)

// fp32 -> bf16 (round to nearest even)
__device__ __forceinline__ unsigned short f2bf(float f) {
    uint32_t x = __builtin_bit_cast(uint32_t, f);
    uint32_t r = x + 0x7FFFu + ((x >> 16) & 1u);
    return (unsigned short)(r >> 16);
}
__device__ __forceinline__ float bf2f(unsigned short u) {
    uint32_t b = ((uint32_t)u) << 16;
    return __builtin_bit_cast(float, b);
}

// ---------- weight prep: fp32 (K x N) -> bf16 transposed (N x K) ----------
__global__ void wprep_kernel(const float* __restrict__ W,
                             unsigned short* __restrict__ Wt,
                             int K, int N) {
    int i = blockIdx.x * blockDim.x + threadIdx.x;   // dst linear index over N*K
    if (i < K * N) {
        int n = i / K;
        int k = i - n * K;
        Wt[i] = f2bf(W[(size_t)k * N + n]);
    }
}

// ---------- bf16 WMMA GEMM: out = (A [+A2]) @ B + bias [+residual] ----------
// A, A2:  fp32, M x 256 (row major).  Bt: bf16 bits, N x 256 (B transposed).
// Block: 256 threads = 8 waves; block tile 128(M) x 128(N); wave tile 32x64.
// A tile (full K) staged via VGPR (needs fp32->bf16 convert + optional add);
// B tile (already bf16) staged via GLOBAL_LOAD_ASYNC_TO_LDS_B128 (ASYNCcnt).
__global__ __launch_bounds__(256) void gemm_bf16_wmma(
    const float* __restrict__ A, const float* __restrict__ A2,
    const unsigned short* __restrict__ Bt,
    const float* __restrict__ bias,
    const float* __restrict__ residual,
    float* __restrict__ outF, unsigned short* __restrict__ outBF,
    int M, int N)
{
    __shared__ alignas(16) unsigned short ldsA[128 * LDSPITCH]; // ~66 KB
    __shared__ alignas(16) unsigned short ldsB[128 * LDSPITCH]; // ~66 KB

    const int tid   = threadIdx.x;
    const int wave  = tid >> 5;
    const int lane  = tid & 31;
    const int hf    = lane >> 4;     // half of wave
    const int l15   = lane & 15;
    const int waveM = wave & 3;      // 0..3 -> 32-row slice
    const int waveN = wave >> 2;     // 0..1 -> 64-col slice
    const int blockM = blockIdx.x * 128;
    const int blockN = blockIdx.y * 128;

    // ---- prologue: stage A (convert fp32->bf16) and B (async copy) into LDS ----
    {
        // A: 128 rows x 256 cols fp32; 2 threads per row, 128 floats each
        const int arow  = tid >> 1;
        const int acol0 = (tid & 1) * 128;
        const float* as = A + (size_t)(blockM + arow) * KDIM + acol0;
        unsigned short* ad = &ldsA[arow * LDSPITCH + acol0];
        if (A2) {
            const float* as2 = A2 + (size_t)(blockM + arow) * KDIM + acol0;
            #pragma unroll
            for (int j = 0; j < 32; ++j) {
                float4 f = *(const float4*)(as + j * 4);
                float4 g = *(const float4*)(as2 + j * 4);
                ad[j*4+0] = f2bf(f.x + g.x); ad[j*4+1] = f2bf(f.y + g.y);
                ad[j*4+2] = f2bf(f.z + g.z); ad[j*4+3] = f2bf(f.w + g.w);
            }
        } else {
            #pragma unroll
            for (int j = 0; j < 32; ++j) {
                float4 f = *(const float4*)(as + j * 4);
                ad[j*4+0] = f2bf(f.x); ad[j*4+1] = f2bf(f.y);
                ad[j*4+2] = f2bf(f.z); ad[j*4+3] = f2bf(f.w);
            }
        }
        // B: 128 rows x 256 bf16 = 4096 16B chunks; 16 async copies per thread
        #pragma unroll
        for (int i = 0; i < 16; ++i) {
            const int c   = tid + 256 * i;
            const int row = c >> 5;         // 0..127
            const int k16 = c & 31;         // 16B chunk within row
            unsigned long long g =
                (unsigned long long)(uintptr_t)(Bt + (size_t)(blockN + row) * KDIM + k16 * 8);
            unsigned l = (unsigned)(uintptr_t)(&ldsB[row * LDSPITCH + k16 * 8]);
            asm volatile("global_load_async_to_lds_b128 %0, %1, off"
                         :: "v"(l), "v"(g) : "memory");
        }
        asm volatile("s_wait_asynccnt 0x0" ::: "memory");
    }
    __syncthreads();

    // ---- main: 8 fully-unrolled K-steps, no barriers (LDS is read-only now) ----
    v8f acc[2][4] = {};
    #pragma unroll
    for (int kk = 0; kk < KDIM; kk += 32) {
        union { v16bf v; uint4 q[2]; } afrag[2], bfrag[4];
        // A frags per ISA 16-bit A layout:
        //   lanes 0-15: K = {kk+0..7, kk+16..23}; lanes 16-31: +8
        #pragma unroll
        for (int mi = 0; mi < 2; ++mi) {
            const unsigned short* rp = &ldsA[(waveM * 32 + mi * 16 + l15) * LDSPITCH + kk];
            afrag[mi].q[0] = *(const uint4*)(rp + 8 * hf);
            afrag[mi].q[1] = *(const uint4*)(rp + 16 + 8 * hf);
        }
        // B frags: col = l15, K matches A: lanes 0-15 K={kk+0..7, kk+16..23}; lanes 16-31: +8
        #pragma unroll
        for (int ni = 0; ni < 4; ++ni) {
            const unsigned short* bp =
                &ldsB[(waveN * 64 + ni * 16 + l15) * LDSPITCH + kk + 8 * hf];
            bfrag[ni].q[0] = *(const uint4*)(bp);
            bfrag[ni].q[1] = *(const uint4*)(bp + 16);
        }
        #pragma unroll
        for (int mi = 0; mi < 2; ++mi)
            #pragma unroll
            for (int ni = 0; ni < 4; ++ni)
                acc[mi][ni] = __builtin_amdgcn_wmma_f32_16x16x32_bf16(
                    false, afrag[mi].v, false, bfrag[ni].v,
                    (short)0, acc[mi][ni], false, false);
    }

    // ---- epilogue: C/D layout -> VGPR r holds M = r + 8*hf, col = l15 ----
    #pragma unroll
    for (int mi = 0; mi < 2; ++mi) {
        #pragma unroll
        for (int ni = 0; ni < 4; ++ni) {
            const int nG = blockN + waveN * 64 + ni * 16 + l15;
            float bv = bias ? bias[nG] : 0.0f;
            #pragma unroll
            for (int r = 0; r < 8; ++r) {
                const int mG = blockM + waveM * 32 + mi * 16 + r + 8 * hf;
                float v = acc[mi][ni][r] + bv;
                if (residual) v += residual[(size_t)mG * N + nG];
                if (outF) outF[(size_t)mG * N + nG] = v;
                else      outBF[(size_t)mG * N + nG] = f2bf(v);
            }
        }
    }
}

// ---------- deformable sampling: one wave per (query, head), lane = channel ----------
__global__ __launch_bounds__(256) void sample_kernel(
    const unsigned short* __restrict__ vproj,   // bf16, (L*NQ, C)
    const float* __restrict__ offBuf,           // (NQ, 256) = (H, L, P, 2)
    const float* __restrict__ attnLog,          // (NQ, 128) = (H, L*P)
    const float* __restrict__ refPts,           // (NQ, L, 2)
    float* __restrict__ sampled)                // (NQ, C) fp32
{
    const int wave = threadIdx.x >> 5;
    const int d    = threadIdx.x & 31;           // channel within head (DH==32)
    const int idx  = blockIdx.x * 8 + wave;      // q * HEADS + h
    const int q    = idx >> 3;
    const int h    = idx & 7;

    // softmax over L*P = 16 logits (redundant per lane; same scalars across wave)
    const float* lg = attnLog + (size_t)q * (HEADS * LEVELS * POINTS) + h * (LEVELS * POINTS);
    float w[16];
    float mx = -1e30f;
    #pragma unroll
    for (int i = 0; i < 16; ++i) { w[i] = lg[i]; mx = fmaxf(mx, w[i]); }
    float s = 0.0f;
    #pragma unroll
    for (int i = 0; i < 16; ++i) { w[i] = __expf(w[i] - mx); s += w[i]; }
    const float inv = 1.0f / s;

    const float* ofq = offBuf + (size_t)q * 256;
    float acc = 0.0f;

    #pragma unroll
    for (int l = 0; l < LEVELS; ++l) {
        const float rx = refPts[((size_t)q * LEVELS + l) * 2 + 0];
        const float ry = refPts[((size_t)q * LEVELS + l) * 2 + 1];
        const unsigned short* vl = vproj + (size_t)l * NQ * C_DIM + h * DH + d;
        #pragma unroll
        for (int p = 0; p < POINTS; ++p) {
            const int oc = ((h * LEVELS + l) * POINTS + p) * 2;
            // loc_norm * W - 0.5, with offset/normalizer folded (norm == W == H == 128)
            const float x = rx * (float)WGRID + ofq[oc + 0] - 0.5f;
            const float y = ry * (float)HGRID + ofq[oc + 1] - 0.5f;
            const float x0f = floorf(x), y0f = floorf(y);
            const int x0 = (int)x0f, y0 = (int)y0f;
            const float wx1 = x - x0f, wy1 = y - y0f;
            const float wx0 = 1.0f - wx1, wy0 = 1.0f - wy1;
            const float aw = w[l * POINTS + p] * inv;

            float v00 = 0.f, v01 = 0.f, v10 = 0.f, v11 = 0.f;
            if ((unsigned)y0 < (unsigned)HGRID) {
                const unsigned short* row = vl + (size_t)(y0 * WGRID) * C_DIM;
                if ((unsigned)x0       < (unsigned)WGRID) v00 = bf2f(row[(size_t)x0 * C_DIM]);
                if ((unsigned)(x0 + 1) < (unsigned)WGRID) v01 = bf2f(row[(size_t)(x0 + 1) * C_DIM]);
            }
            if ((unsigned)(y0 + 1) < (unsigned)HGRID) {
                const unsigned short* row = vl + (size_t)((y0 + 1) * WGRID) * C_DIM;
                if ((unsigned)x0       < (unsigned)WGRID) v10 = bf2f(row[(size_t)x0 * C_DIM]);
                if ((unsigned)(x0 + 1) < (unsigned)WGRID) v11 = bf2f(row[(size_t)(x0 + 1) * C_DIM]);
            }
            acc += aw * (wy0 * (wx0 * v00 + wx1 * v01) + wy1 * (wx0 * v10 + wx1 * v11));
        }
    }
    sampled[(size_t)q * C_DIM + h * DH + d] = acc;
}

// ---------- host ----------
extern "C" void kernel_launch(void* const* d_in, const int* in_sizes, int n_in,
                              void* d_out, int out_size, void* d_ws, size_t ws_size,
                              hipStream_t stream) {
    const float* query  = (const float*)d_in[0];
    const float* qpos   = (const float*)d_in[1];
    const float* value  = (const float*)d_in[2];
    const float* refp   = (const float*)d_in[3];
    // d_in[4] spatial_shapes: static config (128,128) — folded into constants
    const float* W_off  = (const float*)d_in[5];
    const float* b_off  = (const float*)d_in[6];
    const float* W_attn = (const float*)d_in[7];
    const float* b_attn = (const float*)d_in[8];
    const float* W_val  = (const float*)d_in[9];
    const float* b_val  = (const float*)d_in[10];
    const float* W_out  = (const float*)d_in[11];
    const float* b_out  = (const float*)d_in[12];

    // workspace carve-up (all region sizes are multiples of 64KB -> aligned)
    char* p = (char*)d_ws;
    unsigned short* WvalT  = (unsigned short*)p; p += (size_t)256 * 256 * 2;
    unsigned short* WoffT  = (unsigned short*)p; p += (size_t)256 * 256 * 2;
    unsigned short* WattnT = (unsigned short*)p; p += (size_t)128 * 256 * 2;
    unsigned short* WoutT  = (unsigned short*)p; p += (size_t)256 * 256 * 2;
    unsigned short* vproj  = (unsigned short*)p; p += (size_t)LEVELS * NQ * C_DIM * 2; // bf16
    float*          offBuf = (float*)p;          p += (size_t)NQ * 256 * 4;
    float*          attnLg = (float*)p;          p += (size_t)NQ * 128 * 4;
    float*          sampled= (float*)p;          p += (size_t)NQ * C_DIM * 4;

    // 1) weight prep (bf16, transposed to N x K)
    wprep_kernel<<<(256 * 256 + 255) / 256, 256, 0, stream>>>(W_val,  WvalT,  256, 256);
    wprep_kernel<<<(256 * 256 + 255) / 256, 256, 0, stream>>>(W_off,  WoffT,  256, 256);
    wprep_kernel<<<(256 * 128 + 255) / 256, 256, 0, stream>>>(W_attn, WattnT, 256, 128);
    wprep_kernel<<<(256 * 256 + 255) / 256, 256, 0, stream>>>(W_out,  WoutT,  256, 256);

    // 2) value projection: (L*NQ, C) = value @ W_val + b_val  -> bf16
    {
        dim3 g(LEVELS * NQ / 128, C_DIM / 128);
        gemm_bf16_wmma<<<g, 256, 0, stream>>>(value, nullptr, WvalT, b_val,
                                              nullptr, nullptr, vproj,
                                              LEVELS * NQ, C_DIM);
    }
    // 3) sampling offsets: (NQ, 256) = (query+query_pos) @ W_off + b_off -> fp32
    {
        dim3 g(NQ / 128, 256 / 128);
        gemm_bf16_wmma<<<g, 256, 0, stream>>>(query, qpos, WoffT, b_off,
                                              nullptr, offBuf, nullptr,
                                              NQ, 256);
    }
    // 4) attention logits: (NQ, 128) = (query+query_pos) @ W_attn + b_attn -> fp32
    {
        dim3 g(NQ / 128, 128 / 128);
        gemm_bf16_wmma<<<g, 256, 0, stream>>>(query, qpos, WattnT, b_attn,
                                              nullptr, attnLg, nullptr,
                                              NQ, 128);
    }
    // 5) bilinear sampling + attention-weighted sum -> sampled (NQ, C) fp32
    sample_kernel<<<NQ * HEADS / 8, 256, 0, stream>>>(vproj, offBuf, attnLg, refp, sampled);

    // 6) output projection + bias + residual(identity=query) -> d_out fp32
    {
        dim3 g(NQ / 128, C_DIM / 128);
        gemm_bf16_wmma<<<g, 256, 0, stream>>>(sampled, nullptr, WoutT, b_out,
                                              query, (float*)d_out, nullptr,
                                              NQ, C_DIM);
    }
}
